// TGCNmodel_10591389352454
// MI455X (gfx1250) — matrix operand order repaired and checked by reading, not compiled, not run on hardware
//
#include <hip/hip_runtime.h>
#include <cstdint>
#include <cstddef>

#define N_NODES 1024
#define E_EDGES 4096
#define NE      5120      // E + N self-loops
#define NHEADS  8
#define BATCH   32
#define TSTEPS  64
#define HL      512
#define HR      64

typedef __attribute__((ext_vector_type(16))) _Float16 v16h;
typedef __attribute__((ext_vector_type(8)))  float    v8f;

__device__ __forceinline__ float sigm(float x) { return 1.f / (1.f + expf(-x)); }
__device__ __forceinline__ float leaky(float x, float a) { return x > 0.f ? x : a * x; }

// ---------------------------------------------------------------------------
// WMMA wave-level GEMM tile, A staged in LDS (f32), B = f16 weights in
// natural (Nout, K) row-major layout:  C[m][n] = sum_k A[m][k] * Bf[n][k].
// 16-bit A 16x32 lane layout: lane m=lane&15, half=lane>>4,
//   elem i -> k = i + half*16 (one contiguous 16)
// 16-bit B 32x16 lane layout: elem i -> k = i + half*16 (one contiguous 16)
// C/D: vgpr v -> row m0 + v + 8*half, col n0 + (lane&15)
// ---------------------------------------------------------------------------
__device__ __forceinline__ v8f wave_gemm_lds(const float* __restrict__ sA, int ldsa,
                                             const _Float16* __restrict__ Bf, int ldbK,
                                             int n0, int kbase, int Kc, v8f acc)
{
  const int lane = threadIdx.x & 31;
  const int lm   = lane & 15;
  const int half = lane >> 4;
  const float*    ar = sA + lm * ldsa;
  const _Float16* br = Bf + (size_t)(n0 + lm) * ldbK + kbase;
  for (int kk = 0; kk < Kc; kk += 32) {
    v16h a, b;
#pragma unroll
    for (int i = 0; i < 16; ++i) {
      a[i] = (_Float16)ar[kk + half * 16 + i];
      b[i] = br[kk + half * 16 + i];
    }
    acc = __builtin_amdgcn_wmma_f32_16x16x32_f16(false, a, false, b,
                                                 (short)0, acc, false, false);
  }
  return acc;
}

// ---------------- one-time f32 -> f16 weight conversion --------------------
__global__ void k_cvt(const float* __restrict__ W, _Float16* __restrict__ o, int n) {
  int i = blockIdx.x * blockDim.x + threadIdx.x;
  if (i < n) o[i] = (_Float16)W[i];
}

__global__ void k_zero(float* p, int n) {
  int i = blockIdx.x * blockDim.x + threadIdx.x;
  if (i < n) p[i] = 0.f;
}

// ---------------- deterministic CSR (edges grouped by dst) -----------------
__global__ __launch_bounds__(1024) void k_csr_ptr(const int* __restrict__ ei,
                                                  int* __restrict__ csr_ptr)
{
  __shared__ int sdeg[N_NODES];
  int n = threadIdx.x;                       // one block of 1024 == N_NODES
  int deg = 1;                               // self loop
  for (int e = 0; e < E_EDGES; ++e) deg += (ei[E_EDGES + e] == n) ? 1 : 0;
  sdeg[n] = deg;
  __syncthreads();
  if (n == 0) {                              // serial exclusive scan (1024)
    int acc = 0;
    for (int i = 0; i < N_NODES; ++i) { int d = sdeg[i]; csr_ptr[i] = acc; acc += d; }
    csr_ptr[N_NODES] = acc;                  // == NE
  }
}

__global__ void k_csr_fill(const int* __restrict__ ei, const int* __restrict__ csr_ptr,
                           int* __restrict__ csr_eid)
{
  int e = blockIdx.x * blockDim.x + threadIdx.x;
  if (e >= NE) return;
  int d = (e < E_EDGES) ? ei[E_EDGES + e] : (e - E_EDGES);
  int rank = 0;                              // deterministic: order by edge id
  for (int q = 0; q < e; ++q) {
    int dq = (q < E_EDGES) ? ei[E_EDGES + q] : (q - E_EDGES);
    rank += (dq == d) ? 1 : 0;
  }
  csr_eid[csr_ptr[d] + rank] = e;
}

// ---------------- phase 1: small LSTM over rainfall ------------------------
__global__ __launch_bounds__(1024) void k_lstm1(
    const float* __restrict__ rain, const float* __restrict__ Wih,
    const float* __restrict__ Whh,  const float* __restrict__ bih,
    const float* __restrict__ bhh,  float* __restrict__ hs)
{
  __shared__ float sh[BATCH * HR], sc[BATCH * HR], sg[BATCH * 4 * HR];
  __shared__ float swi[4 * HR], sb[4 * HR];
  int tid = threadIdx.x;
  if (tid < 4 * HR) { swi[tid] = Wih[tid]; sb[tid] = bih[tid] + bhh[tid]; }
  for (int i = tid; i < BATCH * HR; i += 1024) { sh[i] = 0.f; sc[i] = 0.f; }
  __syncthreads();
  for (int t = 0; t < TSTEPS; ++t) {
    for (int q = 0; q < 8; ++q) {            // 8192 gate outputs
      int idx = tid + q * 1024;
      int b = idx >> 8, j = idx & 255;
      float acc = rain[b * TSTEPS + t] * swi[j] + sb[j];
      const float* wr = Whh + j * HR;
      const float* hb = sh + b * HR;
#pragma unroll 8
      for (int k = 0; k < HR; ++k) acc += hb[k] * wr[k];
      sg[idx] = acc;
    }
    __syncthreads();
    if (tid < BATCH * HR) {
      int b = tid >> 6, k = tid & 63;
      float gi = sg[b * 256 + k],       gf = sg[b * 256 + 64 + k];
      float gg = sg[b * 256 + 128 + k], go = sg[b * 256 + 192 + k];
      float c = sigm(gf) * sc[tid] + sigm(gi) * tanhf(gg);
      sc[tid] = c;
      float h = sigm(go) * tanhf(c);
      sh[tid] = h;
      hs[((size_t)t * BATCH + b) * HR + k] = h;
    }
    __syncthreads();
  }
}

// ------- runoff = leaky(hs @ fc_W^T + fc_b) + inflow@node753 (fused) -------
__global__ __launch_bounds__(256) void k_runoff_gemm(
    const float* __restrict__ hs, const _Float16* __restrict__ fcH,
    const float* __restrict__ fcb, const float* __restrict__ inflow,
    float* __restrict__ runoff)
{
  __shared__ float sA[16 * HR];
  const int tid = threadIdx.x, wave = tid >> 5;
  const int m0 = blockIdx.x * 16;
  const int n0 = (blockIdx.y * 8 + wave) * 16;
  for (int i = tid; i < 16 * HR; i += 256) {
    int r = i >> 6, c = i & 63;
    sA[i] = hs[(size_t)(m0 + r) * HR + c];
  }
  __syncthreads();
  v8f acc = {0.f,0.f,0.f,0.f,0.f,0.f,0.f,0.f};
  acc = wave_gemm_lds(sA, HR, fcH, HR, n0, 0, HR, acc);
  const int lm = tid & 15, half = (tid >> 4) & 1;
  const int n = n0 + lm;
  const float bias = fcb[n];
#pragma unroll
  for (int v = 0; v < 8; ++v) {
    int row = m0 + v + 8 * half;             // row = t*B + b
    float r = leaky(acc[v] + bias, 0.01f);
    if (n == 753) r += inflow[(row & 31) * TSTEPS + (row >> 5)];
    runoff[(size_t)row * N_NODES + n] = r;
  }
}

// ---------------- GAT layer 1 node features (h1, attn terms) ---------------
__global__ void k_gat1_prep(const float* __restrict__ xn, const float* __restrict__ runoff,
    int t, const float* __restrict__ g1W, const float* __restrict__ g1as,
    const float* __restrict__ g1ad,
    float* __restrict__ h1, float* __restrict__ s1, float* __restrict__ d1,
    float* __restrict__ latOut)
{
  int idx = blockIdx.x * blockDim.x + threadIdx.x;
  if (idx >= BATCH * N_NODES) return;
  int b = idx >> 10, n = idx & 1023;
  float x0  = xn[b * 2 * N_NODES + 2 * n];
  float x1  = xn[b * 2 * N_NODES + 2 * n + 1];
  float lat = runoff[((size_t)t * BATCH + b) * N_NODES + n];
  latOut[((size_t)b * TSTEPS + t) * N_NODES + n] = lat;
  float hv[24];
#pragma unroll
  for (int j = 0; j < 24; ++j) {
    hv[j] = x0 * g1W[j] + x1 * g1W[24 + j] + lat * g1W[48 + j];
    h1[(size_t)idx * 24 + j] = hv[j];
  }
#pragma unroll
  for (int hd = 0; hd < NHEADS; ++hd) {
    float s = 0.f, d = 0.f;
#pragma unroll
    for (int oc = 0; oc < 3; ++oc) {
      s += hv[hd * 3 + oc] * g1as[hd * 3 + oc];
      d += hv[hd * 3 + oc] * g1ad[hd * 3 + oc];
    }
    s1[(size_t)idx * 8 + hd] = s;
    d1[(size_t)idx * 8 + hd] = d;
  }
}

// -------- GAT1 gather softmax + aggregate + GAT2 node prep (fused) ---------
__global__ void k_gat1_gather(const int* __restrict__ ei, const int* __restrict__ csr_ptr,
    const int* __restrict__ csr_eid,
    const float* __restrict__ h1, const float* __restrict__ s1, const float* __restrict__ d1,
    const float* __restrict__ g1b, const float* __restrict__ g2W,
    const float* __restrict__ g2as, const float* __restrict__ g2ad,
    float* __restrict__ attT, float* __restrict__ h2,
    float* __restrict__ s2, float* __restrict__ d2)
{
  int idx = blockIdx.x * blockDim.x + threadIdx.x;
  if (idx >= BATCH * N_NODES) return;
  int b = idx >> 10, n = idx & 1023;
  int st = csr_ptr[n], en = csr_ptr[n + 1];
  const float* sb = s1 + (size_t)b * N_NODES * 8;
  float dv[8], mx[8], den[8];
#pragma unroll
  for (int hd = 0; hd < 8; ++hd) { dv[hd] = d1[(size_t)idx * 8 + hd]; mx[hd] = -1e30f; den[hd] = 0.f; }
  for (int k = st; k < en; ++k) {            // pass 1: segment max
    int e = csr_eid[k];
    int s_ = (e < E_EDGES) ? ei[e] : n;
#pragma unroll
    for (int hd = 0; hd < 8; ++hd)
      mx[hd] = fmaxf(mx[hd], leaky(sb[s_ * 8 + hd] + dv[hd], 0.2f));
  }
  for (int k = st; k < en; ++k) {            // pass 2: denom
    int e = csr_eid[k];
    int s_ = (e < E_EDGES) ? ei[e] : n;
#pragma unroll
    for (int hd = 0; hd < 8; ++hd)
      den[hd] += expf(leaky(sb[s_ * 8 + hd] + dv[hd], 0.2f) - mx[hd]);
  }
  float out[24];
#pragma unroll
  for (int j = 0; j < 24; ++j) out[j] = 0.f;
  for (int k = st; k < en; ++k) {            // pass 3: alpha + aggregate
    int e = csr_eid[k];
    int s_ = (e < E_EDGES) ? ei[e] : n;
    const float* hsv = h1 + ((size_t)b * N_NODES + s_) * 24;
#pragma unroll
    for (int hd = 0; hd < 8; ++hd) {
      float ev = expf(leaky(sb[s_ * 8 + hd] + dv[hd], 0.2f) - mx[hd]);
      float alpha = ev / (den[hd] + 1e-16f);
      attT[((size_t)b * NE + e) * 8 + hd] = alpha;
#pragma unroll
      for (int oc = 0; oc < 3; ++oc) out[hd * 3 + oc] += hsv[hd * 3 + oc] * alpha;
    }
  }
  // x1 = leaky(out + g1_b), then GAT2 node prep in-register
  float xv[24];
#pragma unroll
  for (int j = 0; j < 24; ++j) xv[j] = leaky(out[j] + g1b[j], 0.01f);
  float h2v[16];
#pragma unroll
  for (int u = 0; u < 16; ++u) {
    float a = 0.f;
#pragma unroll
    for (int j = 0; j < 24; ++j) a += xv[j] * g2W[j * 16 + u];
    h2v[u] = a;
    h2[(size_t)idx * 16 + u] = a;
  }
#pragma unroll
  for (int hd = 0; hd < 8; ++hd) {
    s2[(size_t)idx * 8 + hd] = h2v[hd * 2] * g2as[hd * 2] + h2v[hd * 2 + 1] * g2as[hd * 2 + 1];
    d2[(size_t)idx * 8 + hd] = h2v[hd * 2] * g2ad[hd * 2] + h2v[hd * 2 + 1] * g2ad[hd * 2 + 1];
  }
}

// -------- GAT2 gather softmax + head-mean aggregate -> xf (fused) ----------
__global__ void k_gat2_gather(const int* __restrict__ ei, const int* __restrict__ csr_ptr,
    const int* __restrict__ csr_eid,
    const float* __restrict__ h2, const float* __restrict__ s2, const float* __restrict__ d2,
    const float* __restrict__ g2b, float* __restrict__ xf)
{
  int idx = blockIdx.x * blockDim.x + threadIdx.x;
  if (idx >= BATCH * N_NODES) return;
  int b = idx >> 10, n = idx & 1023;
  int st = csr_ptr[n], en = csr_ptr[n + 1];
  const float* sb = s2 + (size_t)b * N_NODES * 8;
  float dv[8], mx[8], den[8];
#pragma unroll
  for (int hd = 0; hd < 8; ++hd) { dv[hd] = d2[(size_t)idx * 8 + hd]; mx[hd] = -1e30f; den[hd] = 0.f; }
  for (int k = st; k < en; ++k) {
    int e = csr_eid[k];
    int s_ = (e < E_EDGES) ? ei[e] : n;
#pragma unroll
    for (int hd = 0; hd < 8; ++hd)
      mx[hd] = fmaxf(mx[hd], leaky(sb[s_ * 8 + hd] + dv[hd], 0.2f));
  }
  for (int k = st; k < en; ++k) {
    int e = csr_eid[k];
    int s_ = (e < E_EDGES) ? ei[e] : n;
#pragma unroll
    for (int hd = 0; hd < 8; ++hd)
      den[hd] += expf(leaky(sb[s_ * 8 + hd] + dv[hd], 0.2f) - mx[hd]);
  }
  float o0 = 0.f, o1 = 0.f;
  for (int k = st; k < en; ++k) {
    int e = csr_eid[k];
    int s_ = (e < E_EDGES) ? ei[e] : n;
    const float* hsv = h2 + ((size_t)b * N_NODES + s_) * 16;
#pragma unroll
    for (int hd = 0; hd < 8; ++hd) {
      float ev = expf(leaky(sb[s_ * 8 + hd] + dv[hd], 0.2f) - mx[hd]);
      float alpha = ev / (den[hd] + 1e-16f);
      o0 += hsv[hd * 2]     * alpha;
      o1 += hsv[hd * 2 + 1] * alpha;
    }
  }
  o0 *= 0.125f; o1 *= 0.125f;                 // mean over 8 heads
  xf[b * 2 * N_NODES + 2 * n]     = leaky(o0 + g2b[0], 0.01f);
  xf[b * 2 * N_NODES + 2 * n + 1] = leaky(o1 + g2b[1], 0.01f);
}

// ---------------- cell gate GEMM: 8 waves share LDS A-chunks ---------------
__global__ __launch_bounds__(256) void k_cell_gemm(
    const float* __restrict__ xf, const float* __restrict__ hn,
    const _Float16* __restrict__ wihH, const _Float16* __restrict__ whhH,
    const float* __restrict__ bih, const float* __restrict__ bhh, float* __restrict__ g)
{
  __shared__ float sA[16 * 512];              // 32 KB chunk, shared by 8 waves
  const int tid = threadIdx.x, wave = tid >> 5;
  const int m0 = blockIdx.x * 16;
  const int n0 = (blockIdx.y * 8 + wave) * 16;
  v8f acc = {0.f,0.f,0.f,0.f,0.f,0.f,0.f,0.f};
  for (int k0 = 0; k0 < 2 * N_NODES; k0 += 512) {   // xf @ Wih^T
    __syncthreads();
    for (int i = tid; i < 16 * 512; i += 256) {
      int r = i >> 9, c = i & 511;
      sA[i] = xf[(size_t)(m0 + r) * (2 * N_NODES) + k0 + c];
    }
    __syncthreads();
    acc = wave_gemm_lds(sA, 512, wihH, 2 * N_NODES, n0, k0, 512, acc);
  }
  __syncthreads();
  for (int i = tid; i < 16 * 512; i += 256) {       // hn @ Whh^T
    int r = i >> 9, c = i & 511;
    sA[i] = hn[(size_t)(m0 + r) * HL + c];
  }
  __syncthreads();
  acc = wave_gemm_lds(sA, 512, whhH, HL, n0, 0, 512, acc);
  const int lm = tid & 15, half = (tid >> 4) & 1;
  const int n = n0 + lm;
  const float bias = bih[n] + bhh[n];
#pragma unroll
  for (int v = 0; v < 8; ++v)
    g[(size_t)(m0 + v + 8 * half) * (4 * HL) + n] = acc[v] + bias;
}

// ---------------- cell pointwise + layernorm -------------------------------
__global__ __launch_bounds__(512) void k_cell_pw(
    const float* __restrict__ g, float* __restrict__ hn, float* __restrict__ cn,
    const float* __restrict__ lng, const float* __restrict__ lnb)
{
  __shared__ float red[HL];
  int b = blockIdx.x, j = threadIdx.x;
  const float* gb = g + (size_t)b * 2048;
  float gi = gb[j], gf = gb[512 + j], gg = gb[1024 + j], go = gb[1536 + j];
  float c = sigm(gf) * cn[b * HL + j] + sigm(gi) * tanhf(gg);
  cn[b * HL + j] = c;
  float h = sigm(go) * tanhf(c);
  red[j] = h; __syncthreads();
  for (int o = 256; o > 0; o >>= 1) { if (j < o) red[j] += red[j + o]; __syncthreads(); }
  float mu = red[0] * (1.f / HL); __syncthreads();
  float dvv = h - mu;
  red[j] = dvv * dvv; __syncthreads();
  for (int o = 256; o > 0; o >>= 1) { if (j < o) red[j] += red[j + o]; __syncthreads(); }
  float var = red[0] * (1.f / HL);
  hn[b * HL + j] = dvv * rsqrtf(var + 1e-5f) * lng[j] + lnb[j];
}

// ---------------- output linear (WMMA) + softplus --------------------------
__global__ __launch_bounds__(256) void k_lin_gemm(
    const float* __restrict__ hn, const _Float16* __restrict__ linH,
    const float* __restrict__ linb, float* __restrict__ xn,
    float* __restrict__ pred, int t)
{
  __shared__ float sA[16 * HL];
  const int tid = threadIdx.x, wave = tid >> 5;
  const int m0 = blockIdx.x * 16;
  const int n0 = (blockIdx.y * 8 + wave) * 16;
  for (int i = tid; i < 16 * HL; i += 256) {
    int r = i >> 9, c = i & 511;
    sA[i] = hn[(size_t)(m0 + r) * HL + c];
  }
  __syncthreads();
  v8f acc = {0.f,0.f,0.f,0.f,0.f,0.f,0.f,0.f};
  acc = wave_gemm_lds(sA, HL, linH, HL, n0, 0, HL, acc);
  const int lm = tid & 15, half = (tid >> 4) & 1;
  const int n = n0 + lm;
  const float bias = linb[n];
#pragma unroll
  for (int v = 0; v < 8; ++v) {
    int b = m0 + v + 8 * half;
    float x  = acc[v] + bias;
    float sp = (x > 20.f) ? x : log1pf(expf(x));
    xn[(size_t)b * 2 * N_NODES + n] = sp;
    pred[((size_t)b * TSTEPS + t) * 2 * N_NODES + n] = sp;
  }
}

// ===========================================================================
extern "C" void kernel_launch(void* const* d_in, const int* in_sizes, int n_in,
                              void* d_out, int out_size, void* d_ws, size_t ws_size,
                              hipStream_t stream)
{
  (void)in_sizes; (void)n_in; (void)out_size; (void)ws_size;
  const float* rainfall  = (const float*)d_in[0];
  const float* inflow    = (const float*)d_in[1];
  const int*   ei        = (const int*)  d_in[2];
  const float* lstm_Wih  = (const float*)d_in[3];
  const float* lstm_Whh  = (const float*)d_in[4];
  const float* lstm_bih  = (const float*)d_in[5];
  const float* lstm_bhh  = (const float*)d_in[6];
  const float* fc_W      = (const float*)d_in[7];
  const float* fc_b      = (const float*)d_in[8];
  const float* g1_W      = (const float*)d_in[9];
  const float* g1_as     = (const float*)d_in[10];
  const float* g1_ad     = (const float*)d_in[11];
  const float* g1_b      = (const float*)d_in[12];
  const float* g2_W      = (const float*)d_in[13];
  const float* g2_as     = (const float*)d_in[14];
  const float* g2_ad     = (const float*)d_in[15];
  const float* g2_b      = (const float*)d_in[16];
  const float* cell_Wih  = (const float*)d_in[17];
  const float* cell_Whh  = (const float*)d_in[18];
  const float* cell_bih  = (const float*)d_in[19];
  const float* cell_bhh  = (const float*)d_in[20];
  const float* ln_g      = (const float*)d_in[21];
  const float* ln_b      = (const float*)d_in[22];
  const float* lin_W     = (const float*)d_in[23];
  const float* lin_b     = (const float*)d_in[24];

  // -------- workspace carve-up (float units) --------
  float* ws = (float*)d_ws;
  float* hs      = ws; ws += TSTEPS * BATCH * HR;
  float* runoff  = ws; ws += TSTEPS * BATCH * N_NODES;
  _Float16* wihH = (_Float16*)ws; ws += (2048 * 2048) / 2;   // f16 weight images
  _Float16* whhH = (_Float16*)ws; ws += (2048 * 512) / 2;
  _Float16* linH = (_Float16*)ws; ws += (2048 * 512) / 2;
  _Float16* fcH  = (_Float16*)ws; ws += (1024 * 64) / 2;
  float* hn   = ws; ws += BATCH * HL;
  float* cn   = ws; ws += BATCH * HL;
  float* xn   = ws; ws += BATCH * 2 * N_NODES;
  float* xf   = ws; ws += BATCH * 2 * N_NODES;
  float* gbuf = ws; ws += BATCH * 4 * HL;
  float* h1   = ws; ws += BATCH * N_NODES * 24;
  float* s1   = ws; ws += BATCH * N_NODES * 8;
  float* d1   = ws; ws += BATCH * N_NODES * 8;
  float* h2   = ws; ws += BATCH * N_NODES * 16;
  float* s2   = ws; ws += BATCH * N_NODES * 8;
  float* d2   = ws; ws += BATCH * N_NODES * 8;
  int* csr_ptr = (int*)ws; ws += 1088;                       // N+1 (padded)
  int* csr_eid = (int*)ws; ws += NE;

  // -------- output carve-up --------
  float* out    = (float*)d_out;
  float* pred   = out;                                        // (B,T,2N)
  float* latOut = out + (size_t)BATCH * TSTEPS * 2 * N_NODES; // (B,T,N,1)
  float* attOut = latOut + (size_t)BATCH * TSTEPS * N_NODES;  // (T,B,NE,8)

  // -------- one-time: f16 weights + deterministic CSR --------
  k_cvt<<<(2048 * 2048 + 255) / 256, 256, 0, stream>>>(cell_Wih, wihH, 2048 * 2048);
  k_cvt<<<(2048 * 512  + 255) / 256, 256, 0, stream>>>(cell_Whh, whhH, 2048 * 512);
  k_cvt<<<(2048 * 512  + 255) / 256, 256, 0, stream>>>(lin_W,   linH, 2048 * 512);
  k_cvt<<<(1024 * 64   + 255) / 256, 256, 0, stream>>>(fc_W,    fcH,  1024 * 64);
  k_csr_ptr<<<1, 1024, 0, stream>>>(ei, csr_ptr);
  k_csr_fill<<<(NE + 255) / 256, 256, 0, stream>>>(ei, csr_ptr, csr_eid);

  // -------- phase 1 --------
  k_lstm1<<<1, 1024, 0, stream>>>(rainfall, lstm_Wih, lstm_Whh, lstm_bih, lstm_bhh, hs);
  k_runoff_gemm<<<dim3(TSTEPS * BATCH / 16, N_NODES / 128), 256, 0, stream>>>(
      hs, fcH, fc_b, inflow, runoff);

  // zero carried state (hn, cn, xn contiguous)
  k_zero<<<(BATCH * HL * 2 + BATCH * 2 * N_NODES + 255) / 256, 256, 0, stream>>>(
      hn, BATCH * HL * 2 + BATCH * 2 * N_NODES);

  // -------- phase 2: 64 sequential graph-recurrent steps (6 kernels/step) --
  const int NODE_BLOCKS = (BATCH * N_NODES + 255) / 256;
  for (int t = 0; t < TSTEPS; ++t) {
    float* attT = attOut + (size_t)t * BATCH * NE * NHEADS;
    k_gat1_prep<<<NODE_BLOCKS, 256, 0, stream>>>(xn, runoff, t, g1_W, g1_as, g1_ad,
                                                 h1, s1, d1, latOut);
    k_gat1_gather<<<NODE_BLOCKS, 256, 0, stream>>>(ei, csr_ptr, csr_eid, h1, s1, d1,
                                                   g1_b, g2_W, g2_as, g2_ad,
                                                   attT, h2, s2, d2);
    k_gat2_gather<<<NODE_BLOCKS, 256, 0, stream>>>(ei, csr_ptr, csr_eid, h2, s2, d2,
                                                   g2_b, xf);
    k_cell_gemm<<<dim3(BATCH / 16, (4 * HL) / 128), 256, 0, stream>>>(
        xf, hn, wihH, whhH, cell_bih, cell_bhh, gbuf);
    k_cell_pw<<<BATCH, HL, 0, stream>>>(gbuf, hn, cn, ln_g, ln_b);
    k_lin_gemm<<<dim3(BATCH / 16, (2 * N_NODES) / 128), 256, 0, stream>>>(
        hn, linH, lin_b, xn, pred, t);
  }
}
